// GNN_Sup_74612171866332
// MI455X (gfx1250) — matrix operand order repaired and checked by reading, not compiled, not run on hardware
//
#include <hip/hip_runtime.h>

#define NNODES 50000
#define NEDGES 800000
#define NGRAPHS 256
#define DIM 128
#define NLAYERS 3
#define WT_STRIDE 136   // 128 + 8 halves pad: keeps 16B alignment, spreads LDS banks

typedef __attribute__((ext_vector_type(16))) _Float16 v16h;
typedef __attribute__((ext_vector_type(8)))  _Float16 v8h;
typedef __attribute__((ext_vector_type(4)))  _Float16 v4h;
typedef __attribute__((ext_vector_type(8)))  float    v8f;
typedef __attribute__((ext_vector_type(4)))  int      v4i;

__device__ __forceinline__ v16h ld_frag(const _Float16* p0, const _Float16* p1) {
  v8h lo = *(const v8h*)p0;
  v8h hi = *(const v8h*)p1;
  return __builtin_shufflevector(lo, hi, 0,1,2,3,4,5,6,7,8,9,10,11,12,13,14,15);
}

// CDNA5 async global->LDS copy of one 16B chunk (ASYNCcnt-tracked).
__device__ __forceinline__ void async_copy16(const _Float16* g, _Float16* l) {
#if __has_builtin(__builtin_amdgcn_global_load_async_to_lds_b128)
  __builtin_amdgcn_global_load_async_to_lds_b128(
      (__attribute__((address_space(1))) v4i*)(__UINTPTR_TYPE__)(const void*)g,
      (__attribute__((address_space(3))) v4i*)l, 0, 0);
#else
  unsigned lofs = (unsigned)(__UINTPTR_TYPE__)l;   // low 32 bits = LDS offset
  asm volatile("global_load_async_to_lds_b128 %0, %1, off"
               :: "v"(lofs), "v"((unsigned long long)(__UINTPTR_TYPE__)g)
               : "memory");
#endif
}

__device__ __forceinline__ void wait_async0() {
#if __has_builtin(__builtin_amdgcn_s_wait_asynccnt)
  __builtin_amdgcn_s_wait_asynccnt(0);
#else
  asm volatile("s_wait_asynccnt 0" ::: "memory");
#endif
}

__global__ void fzero_kernel(float* __restrict__ p, int n) {
  int i = blockIdx.x * blockDim.x + threadIdx.x;
  if (i < n) p[i] = 0.0f;
}

// One-shot: transpose + convert all weights to f16: wf16[slot][n][k] = W[slot][k][n]
// slots: 0..5 = {W1,W2} x 3 layers (2l -> W1, 2l+1 -> W2), 6 = We
__global__ void wconv_kernel(const float* __restrict__ W1,
                             const float* __restrict__ W2,
                             const float* __restrict__ We,
                             _Float16* __restrict__ out) {
  int i = blockIdx.x * blockDim.x + threadIdx.x;
  if (i >= 7 * DIM * DIM) return;
  int slot = i >> 14;
  int r    = i & 16383;
  int n    = r >> 7;
  int k    = r & 127;
  const float* src;
  if (slot < 6) {
    int l = slot >> 1;
    src = (slot & 1) ? (W2 + (size_t)l * DIM * DIM) : (W1 + (size_t)l * DIM * DIM);
  } else {
    src = We;
  }
  out[i] = (_Float16)src[k * DIM + n];
}

// One wave per edge; each lane handles 4 channels (float4 gather + 4 f32 atomics).
__global__ void scatter_kernel(const float* __restrict__ x,
                               const int* __restrict__ src,
                               const int* __restrict__ dst,
                               float* __restrict__ agg) {
  int t = blockIdx.x * blockDim.x + threadIdx.x;
  int e = t >> 5;
  if (e >= NEDGES) return;
  int q = (t & 31) << 2;
  int s = src[e], d = dst[e];
  const float4 v = *(const float4*)(x + (size_t)s * DIM + q);
  float* p = agg + (size_t)d * DIM + q;
  atomicAdd(p + 0, v.x);
  atomicAdd(p + 1, v.y);
  atomicAdd(p + 2, v.z);
  atomicAdd(p + 3, v.w);
}

// Async-load pre-transposed f16 weight matrix [128][128] into padded LDS [128][WT_STRIDE].
// 2048 16B chunks, 16 per thread; per-lane LDS addresses preserve the pad.
__device__ __forceinline__ void issue_w_async(const _Float16* __restrict__ Wg,
                                              _Float16* __restrict__ Wlds, int tid) {
  #pragma unroll
  for (int i = 0; i < 16; ++i) {
    int chunk = tid + 128 * i;
    int n = chunk >> 4;
    int c = chunk & 15;
    async_copy16(Wg + (size_t)chunk * 8, Wlds + n * WT_STRIDE + c * 8);
  }
}

// Fused GIN MLP: h = relu(relu(((1+eps)x+agg)@W1+b1)@W2+b2), writes pre-BN
// output over agg (rows are block-private) and accumulates BN sum/sumsq.
// Block = 128 threads = 4 waves; tile = 16 rows x 128 cols; wave w owns
// column tiles 2w and 2w+1. WMMA f32<-f16 16x16x32, K loop of 4.
__launch_bounds__(128)
__global__ void gin_mlp_kernel(const float* __restrict__ xin,
                               const float* __restrict__ agg,
                               const _Float16* __restrict__ W1t,
                               const float* __restrict__ b1,
                               const _Float16* __restrict__ W2t,
                               const float* __restrict__ b2,
                               const float* __restrict__ epsp,
                               float* __restrict__ y,
                               float* __restrict__ stats) {
  __shared__ __align__(16) _Float16 Wt[DIM * WT_STRIDE];  // W^T: Wt[n][k]
  __shared__ __align__(16) _Float16 Xs[16 * WT_STRIDE];   // X tile / H tile
  __shared__ float ssum[DIM];
  __shared__ float ssq[DIM];

  const int tid  = threadIdx.x;
  const int wv   = tid >> 5;
  const int lane = tid & 31;
  const int m0   = blockIdx.x * 16;
  const float epsv = 1.0f + epsp[0];

  // Kick off async W1^T -> LDS first; overlaps with X-tile conversion below.
  issue_w_async(W1t, Wt, tid);

  ssum[tid] = 0.0f;
  ssq[tid]  = 0.0f;

  // X tile: Xs[m][k] = (1+eps)*x + agg, float4 -> packed 4xf16 per iteration.
  #pragma unroll
  for (int i = 0; i < 4; ++i) {
    int chunk = tid + 128 * i;              // 512 float4 chunks in 16x128 tile
    int r = chunk >> 5;
    int c = (chunk & 31) << 2;
    const float4 xv = *(const float4*)(xin + (size_t)(m0 + r) * DIM + c);
    const float4 av = *(const float4*)(agg + (size_t)(m0 + r) * DIM + c);
    v4h h;
    h.x = (_Float16)fmaf(epsv, xv.x, av.x);
    h.y = (_Float16)fmaf(epsv, xv.y, av.y);
    h.z = (_Float16)fmaf(epsv, xv.z, av.z);
    h.w = (_Float16)fmaf(epsv, xv.w, av.w);
    *(v4h*)(Xs + r * WT_STRIDE + c) = h;
  }
  wait_async0();
  __syncthreads();

  const int row  = lane & 15;          // A row / B column-in-tile / D column
  const int kh   = (lane >> 4) << 3;   // unused; A-frag also uses kb16 (0 or 16)
  const int kb16 = (lane >> 4) << 4;   // B-frag K phase: 0 or 16
  const int r0   = (lane >> 4) << 3;   // D rows: 0..7 or 8..15
  const int c0   = (wv * 2 + 0) * 16 + row;
  const int c1   = (wv * 2 + 1) * 16 + row;

  // ---- GEMM1 ----
  v8f acc0 = {}, acc1 = {};
  #pragma unroll
  for (int kk = 0; kk < DIM; kk += 32) {
    const _Float16* pa  = Xs + row * WT_STRIDE + kk + kb16;
    v16h a = ld_frag(pa, pa + 8);
    const _Float16* pb0 = Wt + c0 * WT_STRIDE + kk + kb16;
    const _Float16* pb1 = Wt + c1 * WT_STRIDE + kk + kb16;
    acc0 = __builtin_amdgcn_wmma_f32_16x16x32_f16(false, a, false, ld_frag(pb0, pb0 + 8),
                                                  (short)0, acc0, false, false);
    acc1 = __builtin_amdgcn_wmma_f32_16x16x32_f16(false, a, false, ld_frag(pb1, pb1 + 8),
                                                  (short)0, acc1, false, false);
  }
  __syncthreads();

  // All GEMM1 reads of Wt done: start async W2^T reload immediately.
  issue_w_async(W2t, Wt, tid);

  // Epilogue1: H = relu(acc + b1) -> Xs (reuse).
  {
    const float bb0 = b1[c0], bb1 = b1[c1];
    #pragma unroll
    for (int i = 0; i < 8; ++i) {
      Xs[(r0 + i) * WT_STRIDE + c0] = (_Float16)fmaxf(acc0[i] + bb0, 0.0f);
      Xs[(r0 + i) * WT_STRIDE + c1] = (_Float16)fmaxf(acc1[i] + bb1, 0.0f);
    }
  }
  wait_async0();
  __syncthreads();

  // ---- GEMM2 ----
  v8f acc2 = {}, acc3 = {};
  #pragma unroll
  for (int kk = 0; kk < DIM; kk += 32) {
    const _Float16* pa  = Xs + row * WT_STRIDE + kk + kb16;
    v16h a = ld_frag(pa, pa + 8);
    const _Float16* pb0 = Wt + c0 * WT_STRIDE + kk + kb16;
    const _Float16* pb1 = Wt + c1 * WT_STRIDE + kk + kb16;
    acc2 = __builtin_amdgcn_wmma_f32_16x16x32_f16(false, a, false, ld_frag(pb0, pb0 + 8),
                                                  (short)0, acc2, false, false);
    acc3 = __builtin_amdgcn_wmma_f32_16x16x32_f16(false, a, false, ld_frag(pb1, pb1 + 8),
                                                  (short)0, acc3, false, false);
  }

  // Epilogue2: x = relu(h @ W2 + b2); store + BN partial stats
  {
    const float bb0 = b2[c0], bb1 = b2[c1];
    float s0 = 0.f, q0 = 0.f, s1 = 0.f, q1 = 0.f;
    #pragma unroll
    for (int i = 0; i < 8; ++i) {
      float v0 = fmaxf(acc2[i] + bb0, 0.0f);
      float v1 = fmaxf(acc3[i] + bb1, 0.0f);
      y[(size_t)(m0 + r0 + i) * DIM + c0] = v0;
      y[(size_t)(m0 + r0 + i) * DIM + c1] = v1;
      s0 += v0; q0 += v0 * v0;
      s1 += v1; q1 += v1 * v1;
    }
    atomicAdd(&ssum[c0], s0); atomicAdd(&ssq[c0], q0);
    atomicAdd(&ssum[c1], s1); atomicAdd(&ssq[c1], q1);
  }
  __syncthreads();
  atomicAdd(&stats[tid],       ssum[tid]);
  atomicAdd(&stats[DIM + tid], ssq[tid]);
}

__global__ void bn_finalize_kernel(const float* __restrict__ stats,
                                   const float* __restrict__ gamma,
                                   const float* __restrict__ beta,
                                   float* __restrict__ bnp) {
  int c = threadIdx.x;
  float mu  = stats[c] * (1.0f / NNODES);
  float var = stats[DIM + c] * (1.0f / NNODES) - mu * mu;
  float sc  = gamma[c] * rsqrtf(var + 1e-5f);
  bnp[c]       = sc;
  bnp[DIM + c] = beta[c] - mu * sc;
}

__global__ void bn_apply_kernel(const float* __restrict__ y,
                                const float* __restrict__ bnp,
                                float* __restrict__ xout) {
  int i = blockIdx.x * blockDim.x + threadIdx.x;   // over N*D/4 float4s
  if (i < NNODES * DIM / 4) {
    int c = (i & 31) << 2;
    float4 v = ((const float4*)y)[i];
    float4 r;
    r.x = fmaf(v.x, bnp[c + 0], bnp[DIM + c + 0]);
    r.y = fmaf(v.y, bnp[c + 1], bnp[DIM + c + 1]);
    r.z = fmaf(v.z, bnp[c + 2], bnp[DIM + c + 2]);
    r.w = fmaf(v.w, bnp[c + 3], bnp[DIM + c + 3]);
    ((float4*)xout)[i] = r;
  }
}

__global__ void pool_kernel(const float* __restrict__ x,
                            const int* __restrict__ batch,
                            float* __restrict__ pooled,
                            float* __restrict__ counts) {
  int t = blockIdx.x * blockDim.x + threadIdx.x;
  int node = t >> 5;
  if (node >= NNODES) return;
  int q = (t & 31) << 2;
  int g = batch[node];
  const float4 v = *(const float4*)(x + (size_t)node * DIM + q);
  float* p = pooled + (size_t)g * DIM + q;
  atomicAdd(p + 0, v.x);
  atomicAdd(p + 1, v.y);
  atomicAdd(p + 2, v.z);
  atomicAdd(p + 3, v.w);
  if (q == 0) atomicAdd(&counts[g], 1.0f);
}

// out[256,128] = (pooled / counts) @ We + be, same WMMA tiling, 16 blocks.
__launch_bounds__(128)
__global__ void head_gemm_kernel(const float* __restrict__ pooled,
                                 const float* __restrict__ counts,
                                 const _Float16* __restrict__ Wet,
                                 const float* __restrict__ be,
                                 float* __restrict__ out) {
  __shared__ __align__(16) _Float16 Wt[DIM * WT_STRIDE];
  __shared__ __align__(16) _Float16 Xs[16 * WT_STRIDE];
  const int tid  = threadIdx.x;
  const int wv   = tid >> 5;
  const int lane = tid & 31;
  const int m0   = blockIdx.x * 16;

  issue_w_async(Wet, Wt, tid);

  #pragma unroll
  for (int i = 0; i < 16; ++i) {
    float inv = 1.0f / fmaxf(counts[m0 + i], 1.0f);
    Xs[i * WT_STRIDE + tid] = (_Float16)(pooled[(size_t)(m0 + i) * DIM + tid] * inv);
  }
  wait_async0();
  __syncthreads();

  const int row  = lane & 15;
  const int kh   = (lane >> 4) << 3;
  const int kb16 = (lane >> 4) << 4;
  const int r0   = (lane >> 4) << 3;
  const int c0   = (wv * 2 + 0) * 16 + row;
  const int c1   = (wv * 2 + 1) * 16 + row;

  v8f acc0 = {}, acc1 = {};
  #pragma unroll
  for (int kk = 0; kk < DIM; kk += 32) {
    const _Float16* pa  = Xs + row * WT_STRIDE + kk + kb16;
    v16h a = ld_frag(pa, pa + 8);
    const _Float16* pb0 = Wt + c0 * WT_STRIDE + kk + kb16;
    const _Float16* pb1 = Wt + c1 * WT_STRIDE + kk + kb16;
    acc0 = __builtin_amdgcn_wmma_f32_16x16x32_f16(false, a, false, ld_frag(pb0, pb0 + 8),
                                                  (short)0, acc0, false, false);
    acc1 = __builtin_amdgcn_wmma_f32_16x16x32_f16(false, a, false, ld_frag(pb1, pb1 + 8),
                                                  (short)0, acc1, false, false);
  }
  const float bb0 = be[c0], bb1 = be[c1];
  #pragma unroll
  for (int i = 0; i < 8; ++i) {
    out[(size_t)(m0 + r0 + i) * DIM + c0] = acc0[i] + bb0;
    out[(size_t)(m0 + r0 + i) * DIM + c1] = acc1[i] + bb1;
  }
}

extern "C" void kernel_launch(void* const* d_in, const int* in_sizes, int n_in,
                              void* d_out, int out_size, void* d_ws, size_t ws_size,
                              hipStream_t stream) {
  (void)in_sizes; (void)n_in; (void)out_size; (void)ws_size;
  const float* x0    = (const float*)d_in[0];
  const int*   ei    = (const int*)d_in[1];
  const int*   batch = (const int*)d_in[2];
  const float* W1    = (const float*)d_in[3];
  const float* b1    = (const float*)d_in[4];
  const float* W2    = (const float*)d_in[5];
  const float* b2    = (const float*)d_in[6];
  const float* eps   = (const float*)d_in[7];
  const float* gamma = (const float*)d_in[8];
  const float* beta  = (const float*)d_in[9];
  const float* We    = (const float*)d_in[10];
  const float* be    = (const float*)d_in[11];
  const int* src = ei;
  const int* dst = ei + NEDGES;

  float*     xA     = (float*)d_ws;                    // [NNODES, D] normalized x
  float*     agg    = xA + (size_t)NNODES * DIM;       // [NNODES, D] agg / pre-BN y
  float*     stats  = agg + (size_t)NNODES * DIM;      // [2, D] sum, sumsq
  float*     bnp    = stats + 2 * DIM;                 // [2, D] scale, shift
  float*     pooled = bnp + 2 * DIM;                   // [NGRAPHS, D]
  float*     counts = pooled + (size_t)NGRAPHS * DIM;  // [NGRAPHS]
  _Float16*  wf16   = (_Float16*)(counts + NGRAPHS);   // [7][128][128] f16 W^T

  // Pre-transpose/convert all weights to f16 once per call.
  wconv_kernel<<<(7 * DIM * DIM + 255) / 256, 256, 0, stream>>>(W1, W2, We, wf16);

  const float* xin = x0;
  for (int l = 0; l < NLAYERS; ++l) {
    fzero_kernel<<<(NNODES * DIM + 255) / 256, 256, 0, stream>>>(agg, NNODES * DIM);
    fzero_kernel<<<1, 256, 0, stream>>>(stats, 2 * DIM);
    scatter_kernel<<<(NEDGES * 32) / 256, 256, 0, stream>>>(xin, src, dst, agg);
    gin_mlp_kernel<<<NNODES / 16, 128, 0, stream>>>(
        xin, agg,
        wf16 + (size_t)(2 * l + 0) * DIM * DIM, b1 + l * DIM,
        wf16 + (size_t)(2 * l + 1) * DIM * DIM, b2 + l * DIM,
        eps + l, agg /* pre-BN out, in place */, stats);
    bn_finalize_kernel<<<1, DIM, 0, stream>>>(stats, gamma + l * DIM, beta + l * DIM, bnp);
    bn_apply_kernel<<<(NNODES * DIM / 4 + 255) / 256, 256, 0, stream>>>(agg, bnp, xA);
    xin = xA;
  }
  fzero_kernel<<<(NGRAPHS * DIM + NGRAPHS + 255) / 256, 256, 0, stream>>>(
      pooled, NGRAPHS * DIM + NGRAPHS);
  pool_kernel<<<(NNODES * 32 + 255) / 256, 256, 0, stream>>>(xA, batch, pooled, counts);
  head_gemm_kernel<<<NGRAPHS / 16, 128, 0, stream>>>(
      pooled, counts, wf16 + (size_t)6 * DIM * DIM, be, (float*)d_out);
}
